// SpeakerRecognitionLoss_29326036697500
// MI455X (gfx1250) — matrix operand order repaired and checked by reading, not compiled, not run on hardware
//
#include <hip/hip_runtime.h>
#include <hip/hip_bf16.h>
#include <math.h>

typedef __attribute__((ext_vector_type(16))) __bf16 v16bf;
typedef __attribute__((ext_vector_type(8)))  __bf16 v8bf;
typedef __attribute__((ext_vector_type(4)))  __bf16 v4bf;
typedef __attribute__((ext_vector_type(8)))  float  v8f;
typedef __attribute__((ext_vector_type(4)))  float  f4;

#define NROWS   4096
#define DDIM    512
#define VDIM    6000
#define MARGIN  0.2f
#define TRIP_W  0.1f
#define MTILES  2              // 16-row WMMA tiles per block => 32 M rows/block
#define MROWS   (16 * MTILES)

static __device__ __forceinline__ v8f wmma_bf16(v16bf a, v16bf b, v8f c) {
  // 8 args: (neg_a, A, neg_b, B, c_mod, C, reuse_a, reuse_b)
  return __builtin_amdgcn_wmma_f32_16x16x32_bf16(false, a, false, b, (short)0, c, false, false);
}

static __device__ __forceinline__ v16bf cat8(v8bf a, v8bf b) {
  return __builtin_shufflevector(a, b, 0,1,2,3,4,5,6,7,8,9,10,11,12,13,14,15);
}

// -------- workspace: acc[4] | sq[4096] | embHi[2M bf16] | embLo[2M bf16] -----
__global__ void init_kernel(float* acc) {
  if (threadIdx.x < 3) acc[threadIdx.x] = 0.0f;
}

// -------- one-shot fp32 -> (hi, lo) bf16 split (hoisted out of GEMM loop) ----
__global__ __launch_bounds__(256) void split_kernel(const float* __restrict__ emb,
                                                    __bf16* __restrict__ hi,
                                                    __bf16* __restrict__ lo) {
  size_t i = ((size_t)blockIdx.x * blockDim.x + threadIdx.x) * 4;
  f4 x = *(const f4*)(emb + i);
  v4bf h, l;
  #pragma unroll
  for (int e = 0; e < 4; ++e) {
    __bf16 he = (__bf16)x[e];          // RNE high part
    h[e] = he;
    l[e] = (__bf16)(x[e] - (float)he); // residual
  }
  *(v4bf*)(hi + i) = h;
  *(v4bf*)(lo + i) = l;
}

// -------- per-row squared norms (from original fp32 for accuracy) ------------
__global__ __launch_bounds__(128) void sq_kernel(const float* __restrict__ emb,
                                                 float* __restrict__ sq) {
  __shared__ float red[128];
  const int row = blockIdx.x, tid = threadIdx.x;
  const float* e = emb + (size_t)row * DDIM;
  float s = 0.0f;
  for (int j = tid; j < DDIM; j += 128) { float v = e[j]; s += v * v; }
  red[tid] = s; __syncthreads();
  for (int st = 64; st > 0; st >>= 1) {
    if (tid < st) red[tid] += red[tid + st];
    __syncthreads();
  }
  if (tid == 0) sq[row] = red[0];
}

// -------- cross entropy: one-pass online logsumexp per row -------------------
__global__ __launch_bounds__(256) void ce_kernel(const float* __restrict__ logits,
                                                 const int* __restrict__ labels,
                                                 float* __restrict__ ce_sum) {
  __shared__ float sm[256], ss[256];
  const int row = blockIdx.x, tid = threadIdx.x;
  const float* x = logits + (size_t)row * VDIM;
  float m = -INFINITY, s = 0.0f;
  for (int j = tid; j < VDIM; j += 256) {
    float v = x[j];
    if (v > m) { s = s * expf(m - v) + 1.0f; m = v; }
    else       { s += expf(v - m); }
  }
  sm[tid] = m; ss[tid] = s; __syncthreads();
  for (int st = 128; st > 0; st >>= 1) {
    if (tid < st) {
      float m1 = sm[tid], s1 = ss[tid];
      float m2 = sm[tid + st], s2 = ss[tid + st];
      float M = fmaxf(m1, m2);
      sm[tid] = M;
      ss[tid] = s1 * expf(m1 - M) + s2 * expf(m2 - M);
    }
    __syncthreads();
  }
  if (tid == 0) {
    float lse = sm[0] + logf(ss[0]);
    int lab = labels[row];
    atomicAdd(ce_sum, lse - x[lab]);
  }
}

// -------- fused Gram-GEMM (LDS-staged A, bf16-split WMMA) + triplet ----------
// block b owns M rows [32b, 32b+32). A panel (hi+lo, 64 KB) is staged in LDS
// once and shared by all 8 waves; waves sweep the 256 N-tiles stride-8.
__global__ __launch_bounds__(256) void triplet_kernel(const __bf16* __restrict__ ehi,
                                                      const __bf16* __restrict__ elo,
                                                      const int* __restrict__ labels,
                                                      const float* __restrict__ sq,
                                                      float* __restrict__ trip_sum,
                                                      float* __restrict__ n_valid) {
  __shared__ __align__(16) __bf16 sAh[MROWS * DDIM];   // 32 KB
  __shared__ __align__(16) __bf16 sAl[MROWS * DDIM];   // 32 KB
  __shared__ int sPos[MROWS], sNeg[MROWS];

  const int tid  = threadIdx.x;
  const int wave = tid >> 5, lane = tid & 31;
  const int half = lane >> 4, l16 = lane & 15;
  const int mBase = blockIdx.x * MROWS;

  if (tid < MROWS) {
    sPos[tid] = __float_as_int(-1.0f);
    sNeg[tid] = __float_as_int(INFINITY);
  }
  // cooperative stage of the A panel (hi+lo) into LDS, 16B chunks
  {
    const uint4* gh = (const uint4*)(ehi + (size_t)mBase * DDIM);
    const uint4* gl = (const uint4*)(elo + (size_t)mBase * DDIM);
    uint4* shd = (uint4*)sAh;
    uint4* sld = (uint4*)sAl;
    #pragma unroll
    for (int i = 0; i < (MROWS * DDIM) / 8 / 256; ++i) {
      int idx = tid + i * 256;
      shd[idx] = gh[idx];
      sld[idx] = gl[idx];
    }
  }
  __syncthreads();

  // per-lane invariants: slot (t, r, lane) -> M = mBase + 16t + r + 8*half
  float sqM[MTILES][8]; int labM[MTILES][8];
  float posM[MTILES][8], negM[MTILES][8];
  #pragma unroll
  for (int t = 0; t < MTILES; ++t)
    #pragma unroll
    for (int r = 0; r < 8; ++r) {
      int M = mBase + 16 * t + r + 8 * half;
      sqM[t][r] = sq[M]; labM[t][r] = labels[M];
      posM[t][r] = -1.0f; negM[t][r] = INFINITY;
    }

  // A fragment base offsets inside LDS (M = lane%16 per tile, half's K window)
  const __bf16* ah[MTILES]; const __bf16* al[MTILES];
  #pragma unroll
  for (int t = 0; t < MTILES; ++t) {
    int off = (16 * t + l16) * DDIM + 16 * half;
    ah[t] = sAh + off; al[t] = sAl + off;
  }

  for (int nt = wave; nt < NROWS / 16; nt += 8) {  // uniform per wave: EXEC all-1s at WMMA
    const int   b_row = nt * 16 + l16;             // B = emb^T: column n = emb row n
    const float sqN   = sq[b_row];
    const int   labN  = labels[b_row];
    const __bf16* bh  = ehi + (size_t)b_row * DDIM + 16 * half;
    const __bf16* bl  = elo + (size_t)b_row * DDIM + 16 * half;

    v8f c[MTILES];
    #pragma unroll
    for (int t = 0; t < MTILES; ++t) c[t] = (v8f){0.f,0.f,0.f,0.f,0.f,0.f,0.f,0.f};

    #pragma unroll 2
    for (int k0 = 0; k0 < DDIM; k0 += 32) {
      // B fragment: K = k0 + 16*half + e, e = 0..15 (contiguous, from L2)
      v16bf bhi = cat8(*(const v8bf*)(bh + k0), *(const v8bf*)(bh + k0 + 8));
      v16bf blo = cat8(*(const v8bf*)(bl + k0), *(const v8bf*)(bl + k0 + 8));
      #pragma unroll
      for (int t = 0; t < MTILES; ++t) {
        // A fragment from LDS: K = k0 + 16*half + e, e = 0..15 (contiguous)
        v16bf ahi = cat8(*(const v8bf*)(ah[t] + k0), *(const v8bf*)(ah[t] + k0 + 8));
        v16bf alo = cat8(*(const v8bf*)(al[t] + k0), *(const v8bf*)(al[t] + k0 + 8));
        // fp32-accurate split GEMM: hi*hi + hi*lo + lo*hi
        c[t] = wmma_bf16(ahi, bhi, c[t]);
        c[t] = wmma_bf16(ahi, blo, c[t]);
        c[t] = wmma_bf16(alo, bhi, c[t]);
      }
    }

    // dist tiles -> masked running max/min (Gram never leaves registers)
    #pragma unroll
    for (int t = 0; t < MTILES; ++t)
      #pragma unroll
      for (int r = 0; r < 8; ++r) {
        int M = mBase + 16 * t + r + 8 * half;
        float d2 = sqM[t][r] + sqN - 2.0f * c[t][r];
        float d  = sqrtf(fmaxf(d2, 0.0f));
        bool same = (labM[t][r] == labN);
        if (same && (M != b_row)) posM[t][r] = fmaxf(posM[t][r], d);
        if (!same)                negM[t][r] = fminf(negM[t][r], d);
      }
  }

  // reduce across the 16 lanes of each half (offsets 8..1 stay within a half)
  #pragma unroll
  for (int t = 0; t < MTILES; ++t)
    #pragma unroll
    for (int r = 0; r < 8; ++r) {
      #pragma unroll
      for (int off = 8; off >= 1; off >>= 1) {
        posM[t][r] = fmaxf(posM[t][r], __shfl_xor(posM[t][r], off, 32));
        negM[t][r] = fminf(negM[t][r], __shfl_xor(negM[t][r], off, 32));
      }
      if (l16 == 0) {
        // distances >= 0 (and -1/+inf sentinels) order correctly as int bits
        atomicMax(&sPos[16 * t + r + 8 * half], __float_as_int(posM[t][r]));
        atomicMin(&sNeg[16 * t + r + 8 * half], __float_as_int(negM[t][r]));
      }
    }
  __syncthreads();

  if (tid < MROWS) {
    float hp = __int_as_float(sPos[tid]);
    float hn = __int_as_float(sNeg[tid]);
    bool valid = (hp >= 0.0f) && (hn < INFINITY);
    if (valid) {
      float pr = fmaxf(hp - hn + MARGIN, 0.0f);
      atomicAdd(trip_sum, pr);
      atomicAdd(n_valid, 1.0f);
    }
  }
}

// -------- combine scalars ----------------------------------------------------
__global__ void finalize_kernel(const float* __restrict__ acc, float* __restrict__ out) {
  float ce = acc[0] / (float)NROWS;
  float nv = acc[2];
  float trip = (nv > 0.0f) ? (acc[1] / fmaxf(nv, 1.0f)) : 0.0f;
  out[0] = ce + TRIP_W * trip;  // total
  out[1] = ce;
  out[2] = trip;
}

extern "C" void kernel_launch(void* const* d_in, const int* in_sizes, int n_in,
                              void* d_out, int out_size, void* d_ws, size_t ws_size,
                              hipStream_t stream) {
  const float* logits = (const float*)d_in[0];   // [4096, 6000]
  const float* emb    = (const float*)d_in[1];   // [4096, 512]
  const int*   labels = (const int*)d_in[2];     // [4096]
  float* out = (float*)d_out;                    // [total, ce, triplet]

  float*  acc = (float*)d_ws;                    // 3 accumulators (+pad)
  float*  sq  = acc + 4;                         // 4096 squared norms
  __bf16* ehi = (__bf16*)(sq + NROWS);           // 4096*512 bf16 high parts
  __bf16* elo = ehi + (size_t)NROWS * DDIM;      // 4096*512 bf16 residuals

  init_kernel<<<1, 32, 0, stream>>>(acc);
  split_kernel<<<(NROWS * DDIM) / (256 * 4), 256, 0, stream>>>(emb, ehi, elo);
  sq_kernel<<<NROWS, 128, 0, stream>>>(emb, sq);
  ce_kernel<<<NROWS, 256, 0, stream>>>(logits, labels, acc + 0);
  triplet_kernel<<<NROWS / MROWS, 256, 0, stream>>>(ehi, elo, labels, sq,
                                                    acc + 1, acc + 2);
  finalize_kernel<<<1, 1, 0, stream>>>(acc, out);
}
